// LMACL_46686294507598
// MI455X (gfx1250) — matrix operand order repaired and checked by reading, not compiled, not run on hardware
//
#include <hip/hip_runtime.h>
#include <math.h>

#define NU 30000
#define NI 15000
#define NNODE (NU + NI)
#define DD 64
#define NH 4
#define NF 16
#define BB 2048
#define DEG 16
#define LOG_EPS (-18.420680743952367f)
#define NTILES ((NNODE + 15) / 16)

typedef __attribute__((ext_vector_type(16))) _Float16 v16h;
typedef __attribute__((ext_vector_type(8)))  float    v8f;

// ---------------- utility kernels ----------------
__global__ void k_fill_f32(float* p, float v, int n) {
    int i = blockIdx.x * blockDim.x + threadIdx.x;
    if (i < n) p[i] = v;
}
__global__ void k_fill_u32(unsigned* p, unsigned v, int n) {
    int i = blockIdx.x * blockDim.x + threadIdx.x;
    if (i < n) p[i] = v;
}
__global__ void k_copy3(const float* s, float* a, float* b, float* c, int n) {
    int i = blockIdx.x * blockDim.x + threadIdx.x;
    if (i < n) { float v = s[i]; a[i] = v; b[i] = v; c[i] = v; }
}

// ---------------- feat = concat(Eu,Ei) @ W  (WMMA f16, f32 accum, W staged in LDS) ----------------
__global__ __launch_bounds__(256) void k_feat_wmma(const float* __restrict__ Eu,
                                                   const float* __restrict__ Ei,
                                                   const float* __restrict__ W,
                                                   float* __restrict__ feat) {
    __shared__ _Float16 Wl[DD * DD];          // 8 KB of the 320 KB/WGP LDS
    int tid = threadIdx.x;
    for (int i = tid; i < DD * DD; i += 256) Wl[i] = (_Float16)W[i];
    __syncthreads();

    int wav  = tid >> 5;                      // 8 waves per block, one 16-row tile each
    int lane = tid & 31;
    int tile = blockIdx.x * 8 + wav;
    if (tile >= NTILES) return;

    int half = lane >> 4;
    int lm   = lane & 15;
    int row  = tile * 16 + lm;
    int rc   = row < NNODE ? row : NNODE - 1;
    const float* hrow = (rc < NU) ? (Eu + (size_t)rc * DD) : (Ei + (size_t)(rc - NU) * DD);

    // A fragments: 2 k-chunks of 32
    v16h a[2];
#pragma unroll
    for (int kc = 0; kc < 2; kc++) {
        int kb = kc * 32 + half * 8;
#pragma unroll
        for (int t = 0; t < 8; t++) {
            a[kc][t]     = (_Float16)hrow[kb + t];
            a[kc][8 + t] = (_Float16)hrow[kb + 16 + t];
        }
    }
    bool full = (tile * 16 + 16) <= NNODE;    // wave-uniform tail test
#pragma unroll
    for (int nt = 0; nt < 4; nt++) {
        int col = nt * 16 + lm;
        v16h b[2];
#pragma unroll
        for (int kc = 0; kc < 2; kc++) {
#pragma unroll
            for (int t = 0; t < 16; t++) {
                int k = kc * 32 + half * 8 + t + (t >= 8 ? 8 : 0);
                b[kc][t] = Wl[k * DD + col];
            }
        }
        v8f c = {};
        c = __builtin_amdgcn_wmma_f32_16x16x32_f16(false, a[0], false, b[0], (short)0, c, false, false);
        c = __builtin_amdgcn_wmma_f32_16x16x32_f16(false, a[1], false, b[1], (short)0, c, false, false);
        if (full) {
#pragma unroll
            for (int r = 0; r < 8; r++) {
                int rg = tile * 16 + r + half * 8;
                feat[(size_t)rg * DD + col] = c[r];
            }
        } else {
#pragma unroll
            for (int r = 0; r < 8; r++) {
                int rg = tile * 16 + r + half * 8;
                if (rg < NNODE) feat[(size_t)rg * DD + col] = c[r];
            }
        }
    }
}

// ---------------- attention coefficients ----------------
__global__ void k_eler(const float* __restrict__ feat, const float* __restrict__ al,
                       const float* __restrict__ ar, float* el, float* er) {
    int i = blockIdx.x * blockDim.x + threadIdx.x;   // node*NH + h
    if (i >= NNODE * NH) return;
    int v = i >> 2, h = i & 3;
    const float* f = feat + (size_t)v * DD + h * NF;
    float sl = 0.f, sr = 0.f;
#pragma unroll
    for (int t = 0; t < NF; t++) { sl += f[t] * al[h * NF + t]; sr += f[t] * ar[h * NF + t]; }
    el[i] = sl; er[i] = sr;
}

// ---------------- GCN propagation ----------------
__global__ void k_zi_scatter(const float* __restrict__ Eu, const float* __restrict__ vals,
                             const int* __restrict__ uix, const int* __restrict__ iix,
                             float* Zi, int nnz) {
    int e = blockIdx.x * blockDim.x + threadIdx.x;
    if (e >= nnz) return;
    float v = vals[e];
    const float* s = Eu + (size_t)uix[e] * DD;
    float* d = Zi + (size_t)iix[e] * DD;
#pragma unroll 4
    for (int k = 0; k < DD; k++) atomicAdd(&d[k], v * s[k]);
}

__global__ void k_update_Eu(float* Eu, float* sEu, const float* __restrict__ Ei,
                            const float* __restrict__ vals, const int* __restrict__ iix) {
    int i = blockIdx.x * blockDim.x + threadIdx.x;   // u*DD + d
    if (i >= NU * DD) return;
    int u = i / DD, d = i % DD;
    float z = 0.f;
    int base = u * DEG;
#pragma unroll
    for (int t = 0; t < DEG; t++) {
        int e = base + t;
        z += vals[e] * Ei[(size_t)iix[e] * DD + d];
    }
    float x = Eu[i] + z;
    Eu[i] = x; sEu[i] += x;
}

__global__ void k_update_Ei(float* Ei, float* sEi, const float* __restrict__ Zi) {
    int i = blockIdx.x * blockDim.x + threadIdx.x;
    if (i >= NI * DD) return;
    float x = Ei[i] + Zi[i];
    Ei[i] = x; sEi[i] += x;
}

// ---------------- GAT edge softmax (ordered-uint atomic max trick) ----------------
__device__ __forceinline__ unsigned f2ord(float f) {
    unsigned u = __float_as_uint(f);
    return (u & 0x80000000u) ? ~u : (u | 0x80000000u);
}
__device__ __forceinline__ float ord2f(unsigned u) {
    return __uint_as_float((u & 0x80000000u) ? (u & 0x7fffffffu) : ~u);
}
__device__ __forceinline__ float leaky(float x) { return x > 0.f ? x : 0.2f * x; }

__global__ void k_edge_max(const float* __restrict__ el, const float* __restrict__ er,
                           const int* __restrict__ src, const int* __restrict__ dst,
                           unsigned* emax, int ne) {
    int i = blockIdx.x * blockDim.x + threadIdx.x;
    if (i >= ne * NH) return;
    int e = i >> 2, h = i & 3;
    float x = leaky(el[src[e] * NH + h] + er[dst[e] * NH + h]);
    atomicMax(&emax[dst[e] * NH + h], f2ord(x));
}
__global__ void k_edge_den(const float* __restrict__ el, const float* __restrict__ er,
                           const int* __restrict__ src, const int* __restrict__ dst,
                           const unsigned* __restrict__ emax, float* den, int ne) {
    int i = blockIdx.x * blockDim.x + threadIdx.x;
    if (i >= ne * NH) return;
    int e = i >> 2, h = i & 3;
    int d = dst[e];
    float x = leaky(el[src[e] * NH + h] + er[d * NH + h]);
    atomicAdd(&den[d * NH + h], __expf(x - ord2f(emax[d * NH + h])));
}
__global__ void k_edge_agg(const float* __restrict__ el, const float* __restrict__ er,
                           const int* __restrict__ src, const int* __restrict__ dst,
                           const unsigned* __restrict__ emax, const float* __restrict__ den,
                           const float* __restrict__ feat, float* gout, int ne) {
    int i = blockIdx.x * blockDim.x + threadIdx.x;
    if (i >= ne * NH) return;
    int e = i >> 2, h = i & 3;
    int s = src[e], d = dst[e];
    float x = leaky(el[s * NH + h] + er[d * NH + h]);
    float a = __expf(x - ord2f(emax[d * NH + h])) / den[d * NH + h];
    const float* fs = feat + (size_t)s * DD + h * NF;
    float* od = gout + (size_t)d * DD + h * NF;
#pragma unroll
    for (int t = 0; t < NF; t++) atomicAdd(&od[t], a * fs[t]);
}

__global__ void k_accum_G(const float* __restrict__ gout, float* sGu, float* sGi) {
    int i = blockIdx.x * blockDim.x + threadIdx.x;
    if (i >= NNODE * DD) return;
    float x = gout[i];
    float e = x > 0.f ? x : (__expf(x) - 1.f);   // elu
    if (i < NU * DD) sGu[i] += e; else sGi[i - NU * DD] += e;
}

// ---------------- streaming logsumexp GEMM: lse over G[ids] @ E^T / TEMP ----------------
__global__ void k_lse_wmma(const int* __restrict__ ids, const float* __restrict__ G,
                           const float* __restrict__ E, int ncand, float* acc) {
    int wave = blockIdx.x;           // BB/16 waves, one 16-row batch tile each
    int lane = threadIdx.x;
    int half = lane >> 4;
    int lm   = lane & 15;
    int b    = wave * 16 + lm;
    const float* grow = G + (size_t)ids[b] * DD;

    v16h a0, a1;
#pragma unroll
    for (int t = 0; t < 8; t++) {
        int k0 = half * 8 + t;
        a0[t]     = (_Float16)grow[k0];
        a0[8 + t] = (_Float16)grow[16 + k0];
        a1[t]     = (_Float16)grow[32 + k0];
        a1[8 + t] = (_Float16)grow[48 + k0];
    }
    const float invT = 5.0f;   // 1/TEMP
    float m[8], s[8];
#pragma unroll
    for (int r = 0; r < 8; r++) { m[r] = -1e30f; s[r] = 0.f; }

    int ntiles = (ncand + 15) / 16;
    for (int jt = 0; jt < ntiles; jt++) {
        int j  = jt * 16 + lm;
        int jc = j < ncand ? j : ncand - 1;
        const float* erow = E + (size_t)jc * DD;
        // prefetch next tile's rows (L2-resident stream; hides gather latency)
        if (jt + 1 < ntiles) {
            int j2  = jt * 16 + 16 + lm;
            int jc2 = j2 < ncand ? j2 : ncand - 1;
            __builtin_prefetch(E + (size_t)jc2 * DD, 0, 1);
        }
        v16h b0, b1;
#pragma unroll
        for (int t = 0; t < 16; t++) {
            int k = half * 8 + t + (t >= 8 ? 8 : 0);
            b0[t] = (_Float16)erow[k];
            b1[t] = (_Float16)erow[32 + k];
        }
        v8f c = {};
        c = __builtin_amdgcn_wmma_f32_16x16x32_f16(false, a0, false, b0, (short)0, c, false, false);
        c = __builtin_amdgcn_wmma_f32_16x16x32_f16(false, a1, false, b1, (short)0, c, false, false);
        bool valid = j < ncand;
#pragma unroll
        for (int r = 0; r < 8; r++) {
            float x  = valid ? c[r] * invT : -1e30f;
            float mn = fmaxf(m[r], x);
            s[r] = s[r] * __expf(m[r] - mn) + __expf(x - mn);
            m[r] = mn;
        }
    }
    // combine (m,s) across the 16 lanes of each half-wave
#pragma unroll
    for (int off = 1; off < 16; off <<= 1) {
#pragma unroll
        for (int r = 0; r < 8; r++) {
            float mo = __shfl_xor(m[r], off, 32);
            float so = __shfl_xor(s[r], off, 32);
            float mn = fmaxf(m[r], mo);
            s[r] = s[r] * __expf(m[r] - mn) + so * __expf(mo - mn);
            m[r] = mn;
        }
    }
    if (lm == 0) {       // lanes 0 and 16: rows 0..7 and 8..15 of the tile
        float lsum = 0.f;
#pragma unroll
        for (int r = 0; r < 8; r++) {
            float lse = m[r] + __logf(s[r]);
            float hi = fmaxf(lse, LOG_EPS), lo = fminf(lse, LOG_EPS);
            lsum += hi + log1pf(__expf(lo - hi));      // logaddexp(lse, log(1e-8))
        }
        atomicAdd(acc, lsum);
    }
}

// ---------------- scalar loss pieces ----------------
__global__ void k_pos(const float* __restrict__ sGu, const float* __restrict__ sEu,
                      const float* __restrict__ sGi, const float* __restrict__ sEi,
                      const int* __restrict__ uids, const int* __restrict__ iids, float* acc) {
    int b = blockIdx.x * blockDim.x + threadIdx.x;
    if (b >= BB) return;
    const float* g = sGu + (size_t)uids[b] * DD;
    const float* e = sEu + (size_t)uids[b] * DD;
    float du = 0.f;
#pragma unroll 8
    for (int d = 0; d < DD; d++) du += g[d] * e[d];
    const float* g2 = sGi + (size_t)iids[b] * DD;
    const float* e2 = sEi + (size_t)iids[b] * DD;
    float di = 0.f;
#pragma unroll 8
    for (int d = 0; d < DD; d++) di += g2[d] * e2[d];
    atomicAdd(acc, du + di);
}

__global__ void k_bpr(const float* __restrict__ sEu, const float* __restrict__ sEi,
                      const int* __restrict__ uids, const int* __restrict__ pos,
                      const int* __restrict__ neg, float* acc) {
    int b = blockIdx.x * blockDim.x + threadIdx.x;
    if (b >= BB) return;
    const float* u = sEu + (size_t)uids[b] * DD;
    const float* p = sEi + (size_t)pos[b] * DD;
    const float* n = sEi + (size_t)neg[b] * DD;
    float diff = 0.f;
#pragma unroll 8
    for (int d = 0; d < DD; d++) diff += u[d] * (p[d] - n[d]);
    float ls = (diff >= 0.f) ? -log1pf(__expf(-diff)) : (diff - log1pf(__expf(diff)));
    atomicAdd(acc, ls);
}

__global__ void k_sumsq(const float* __restrict__ p, int n, float* acc) {
    int i = blockIdx.x * blockDim.x + threadIdx.x;
    int stride = gridDim.x * blockDim.x;
    float s = 0.f;
    for (int k = i; k < n; k += stride) { float v = p[k]; s += v * v; }
    atomicAdd(acc, s);
}

__global__ void k_final(const float* __restrict__ acc, float* out) {
    float negs   = acc[0] / BB + acc[1] / BB;
    float poss   = (acc[2] / BB) * 5.0f;        // /TEMP
    float loss_s = -poss + negs;
    float loss_r = -acc[3] / BB;
    float loss_reg = 1e-7f * acc[4];
    out[0] = loss_r + 0.2f * loss_s + loss_reg;
    out[1] = loss_r;
    out[2] = 0.2f * loss_s;
}

// ---------------- launch ----------------
static inline int cdiv(long long a, int b) { return (int)((a + b - 1) / b); }

extern "C" void kernel_launch(void* const* d_in, const int* in_sizes, int n_in,
                              void* d_out, int out_size, void* d_ws, size_t ws_size,
                              hipStream_t stream) {
    const float* E_u_0 = (const float*)d_in[0];
    const float* E_i_0 = (const float*)d_in[1];
    const float* vals  = (const float*)d_in[2];
    const float* W     = (const float*)d_in[3];
    const float* al    = (const float*)d_in[4];
    const float* ar    = (const float*)d_in[5];
    const int* u_idx   = (const int*)d_in[6];
    const int* i_idx   = (const int*)d_in[7];
    const int* src     = (const int*)d_in[8];
    const int* dst     = (const int*)d_in[9];
    const int* uids    = (const int*)d_in[10];
    const int* iids    = (const int*)d_in[11];
    const int* pos     = (const int*)d_in[12];
    const int* neg     = (const int*)d_in[13];
    int nnz = in_sizes[6];
    int ne  = in_sizes[8];

    float* p = (float*)d_ws;
    float* Eu  = p; p += (size_t)NU * DD;
    float* Ei  = p; p += (size_t)NI * DD;
    float* sEu = p; p += (size_t)NU * DD;
    float* sEi = p; p += (size_t)NI * DD;
    float* sGu = p; p += (size_t)NU * DD;
    float* sGi = p; p += (size_t)NI * DD;
    float* Zi  = p; p += (size_t)NI * DD;
    float* feat= p; p += (size_t)NNODE * DD;
    float* gout= p; p += (size_t)NNODE * DD;
    float* el  = p; p += (size_t)NNODE * NH;
    float* er  = p; p += (size_t)NNODE * NH;
    float* den = p; p += (size_t)NNODE * NH;
    unsigned* emax = (unsigned*)p; p += (size_t)NNODE * NH;
    float* acc = p;

    const int T = 256;
    // init: E = sumE = sumG = E0 ; zero accumulators
    k_copy3<<<cdiv((long long)NU * DD, T), T, 0, stream>>>(E_u_0, Eu, sEu, sGu, NU * DD);
    k_copy3<<<cdiv((long long)NI * DD, T), T, 0, stream>>>(E_i_0, Ei, sEi, sGi, NI * DD);
    k_fill_f32<<<1, 32, 0, stream>>>(acc, 0.f, 8);

    for (int layer = 0; layer < 2; layer++) {
        // GAT feature transform on old embeddings (8 waves/block, W in LDS)
        k_feat_wmma<<<cdiv(NTILES, 8), 256, 0, stream>>>(Eu, Ei, W, feat);
        k_eler<<<cdiv((long long)NNODE * NH, T), T, 0, stream>>>(feat, al, ar, el, er);
        // GCN propagation (item side needs scatter)
        k_fill_f32<<<cdiv((long long)NI * DD, T), T, 0, stream>>>(Zi, 0.f, NI * DD);
        k_zi_scatter<<<cdiv(nnz, T), T, 0, stream>>>(Eu, vals, u_idx, i_idx, Zi, nnz);
        // edge softmax buffers
        k_fill_u32<<<cdiv((long long)NNODE * NH, T), T, 0, stream>>>(emax, 0x007FFFFFu, NNODE * NH); // ord(-inf)
        k_fill_f32<<<cdiv((long long)NNODE * NH, T), T, 0, stream>>>(den, 0.f, NNODE * NH);
        k_fill_f32<<<cdiv((long long)NNODE * DD, T), T, 0, stream>>>(gout, 0.f, NNODE * DD);
        k_edge_max<<<cdiv((long long)ne * NH, T), T, 0, stream>>>(el, er, src, dst, emax, ne);
        k_edge_den<<<cdiv((long long)ne * NH, T), T, 0, stream>>>(el, er, src, dst, emax, den, ne);
        k_edge_agg<<<cdiv((long long)ne * NH, T), T, 0, stream>>>(el, er, src, dst, emax, den, feat, gout, ne);
        // E updates (Eu reads old Ei -> must precede Ei update)
        k_update_Eu<<<cdiv((long long)NU * DD, T), T, 0, stream>>>(Eu, sEu, Ei, vals, i_idx);
        k_update_Ei<<<cdiv((long long)NI * DD, T), T, 0, stream>>>(Ei, sEi, Zi);
        k_accum_G<<<cdiv((long long)NNODE * DD, T), T, 0, stream>>>(gout, sGu, sGi);
    }

    // streaming LSE GEMMs (acc0 = sum logaddexp user, acc1 = item)
    k_lse_wmma<<<BB / 16, 32, 0, stream>>>(uids, sGu, sEu, NU, acc + 0);
    k_lse_wmma<<<BB / 16, 32, 0, stream>>>(iids, sGi, sEi, NI, acc + 1);
    k_pos<<<cdiv(BB, T), T, 0, stream>>>(sGu, sEu, sGi, sEi, uids, iids, acc + 2);
    k_bpr<<<cdiv(BB, T), T, 0, stream>>>(sEu, sEi, uids, pos, neg, acc + 3);
    k_sumsq<<<256, T, 0, stream>>>(E_u_0, NU * DD, acc + 4);
    k_sumsq<<<256, T, 0, stream>>>(E_i_0, NI * DD, acc + 4);
    k_final<<<1, 1, 0, stream>>>(acc, (float*)d_out);
}
